// SphereNet_4879082848306
// MI455X (gfx1250) — compile-verified, matrix operand order checked by reading, not run on hardware
//
#include <hip/hip_runtime.h>
#include <hip/hip_bf16.h>
#include <stdint.h>

// ---------------- constants ----------------
#define EN 100000   // edges
#define TN 500000   // triplets
#define HD 128      // hidden
#define ID 64       // int emb
#define SBF_K 42
#define TB_K 294

// ---------------- WMMA types / helpers ----------------
typedef __attribute__((ext_vector_type(16))) __bf16 bf16x16;
typedef __attribute__((ext_vector_type(8)))  __bf16 bf16x8;
typedef __attribute__((ext_vector_type(8)))  float  floatx8;

__device__ __forceinline__ floatx8 wmma_bf16(bf16x16 a, bf16x16 b, floatx8 c) {
    // (neg_a, A, neg_b, B, c_mod, C, reuse_a, reuse_b)
    return __builtin_amdgcn_wmma_f32_16x16x32_bf16(false, a, false, b, (short)0, c, false, false);
}

// Load one 16x32 bf16 fragment (A from LDS tile, or B from N-major weights).
// Per ISA 7.12.2 (wave32): lanes 0-15 -> row/col = lane, K = kc+{0..7} and kc+{16..23};
// lanes 16-31 -> row/col = lane-16, K = kc+{8..15} and kc+{24..31}.
__device__ __forceinline__ bf16x16 load_frag(const __bf16* base, int stride, int kc, int lane) {
    int half = lane >> 4;
    int rc   = lane & 15;
    const __bf16* p = base + rc * stride + kc + half * 8;
    bf16x8 lo = *(const bf16x8*)(p);
    bf16x8 hi = *(const bf16x8*)(p + 16);
    return __builtin_shufflevector(lo, hi, 0,1,2,3,4,5,6,7,8,9,10,11,12,13,14,15);
}

// Full-K GEMM tile: A from LDS (row stride 136 bf16), B N-major (stride = KD).
template<int KD>
__device__ __forceinline__ floatx8 gemm_tile(const __bf16* AbLDS, const __bf16* WtBase, int lane) {
    floatx8 c = {};
#pragma unroll
    for (int kc = 0; kc < KD; kc += 32) {
        bf16x16 a = load_frag(AbLDS, 136, kc, lane);
        bf16x16 b = load_frag(WtBase, KD,  kc, lane);
        c = wmma_bf16(a, b, c);
    }
    return c;
}

__device__ __forceinline__ float silu(float x) { return x / (1.f + __expf(-x)); }

// ---------------- prep kernels ----------------
// dst[n*K + k] = (bf16) src[k*N + n]   (N-major bf16 weights for B fragments)
__global__ void __launch_bounds__(256) k_transpose_bf16(const float* __restrict__ src,
                                                        __bf16* __restrict__ dst, int K, int N) {
    int g = blockIdx.x * 256 + threadIdx.x;
    if (g < K * N) {
        int k = g / N, n = g % N;
        dst[n * K + k] = (__bf16)src[g];
    }
}

// Zero-padded N-major bf16 B matrix for the basis projections:
// dst[n*KP + k] = src[k*8 + n] for n<8, k<K; else 0.   (dst is [16 x KP])
__global__ void __launch_bounds__(256) k_pad_w1(const float* __restrict__ src,
                                                __bf16* __restrict__ dst, int K, int KP) {
    int g = blockIdx.x * 256 + threadIdx.x;
    if (g < 16 * KP) {
        int n = g / KP, k = g - n * KP;
        float v = (n < 8 && k < K) ? src[k * 8 + n] : 0.f;
        dst[g] = (__bf16)v;
    }
}

// [8][64] fp32 -> [64][8] fp32 (channel-major second-stage weights)
__global__ void __launch_bounds__(256) k_transpose_f32(const float* __restrict__ src,
                                                       float* __restrict__ dst) {
    int g = blockIdx.x * 256 + threadIdx.x;
    if (g < 512) {
        int h = g / 64, c = g % 64;
        dst[c * 8 + h] = src[g];
    }
}

// M_rbf[6][128] = W_rbf1[6][8] @ W_rbf2[8][128]
__global__ void __launch_bounds__(256) k_mrbf(const float* __restrict__ W1,
                                              const float* __restrict__ W2,
                                              float* __restrict__ M) {
    for (int i = threadIdx.x; i < 6 * 128; i += 256) {
        int j = i / 128, c = i % 128;
        float acc = 0.f;
#pragma unroll
        for (int b = 0; b < 8; ++b) acc += W1[j * 8 + b] * W2[b * 128 + c];
        M[i] = acc;
    }
}

__global__ void __launch_bounds__(256) k_zero(float* __restrict__ p, int n) {
    int g = blockIdx.x * 256 + threadIdx.x;
    if (g < n) p[g] = 0.f;
}

// ---------------- edge front: x_ji, x_kj*rbf, down-projection ----------------
__global__ void __launch_bounds__(256) k_edge_front(
    const float* __restrict__ x1, const float* __restrict__ rbf0, const float* __restrict__ Mrbf,
    const __bf16* __restrict__ Wt_ji, const float* __restrict__ b_ji,
    const __bf16* __restrict__ Wt_kj, const float* __restrict__ b_kj,
    const __bf16* __restrict__ Wt_down,
    float* __restrict__ xji_out, float* __restrict__ xkjd_out)
{
    __shared__ __bf16 Abuf[16 * 136];
    __shared__ __bf16 Abuf2[16 * 136];
    __shared__ float  rbf_s[16 * 132];
    __shared__ float  r0s[96];

    int tid = threadIdx.x;
    int e0  = blockIdx.x * 16;

    if (tid < 96) r0s[tid] = rbf0[e0 * 6 + tid];
    for (int i = tid; i < 16 * 128; i += 256) {
        int row = i >> 7, col = i & 127;
        Abuf[row * 136 + col] = (__bf16)x1[(e0 + row) * 128 + col];
    }
    __syncthreads();
    for (int i = tid; i < 16 * 128; i += 256) {
        int row = i >> 7, col = i & 127;
        float acc = 0.f;
#pragma unroll
        for (int j = 0; j < 6; ++j) acc += r0s[row * 6 + j] * Mrbf[j * 128 + col];
        rbf_s[row * 132 + col] = acc;
    }
    __syncthreads();

    int wave = tid >> 5, lane = tid & 31;
    int n0 = wave * 16;
    int rc = lane & 15, half = lane >> 4;
    int col = n0 + rc;

    floatx8 cji = {}, ckj = {};
#pragma unroll
    for (int kc = 0; kc < 128; kc += 32) {
        bf16x16 a  = load_frag(Abuf, 136, kc, lane);
        bf16x16 bj = load_frag(Wt_ji + n0 * 128, 128, kc, lane);
        cji = wmma_bf16(a, bj, cji);
        bf16x16 bk = load_frag(Wt_kj + n0 * 128, 128, kc, lane);
        ckj = wmma_bf16(a, bk, ckj);
    }
    float bjv = b_ji[col], bkv = b_kj[col];
#pragma unroll
    for (int r = 0; r < 8; ++r) {
        int row = r + half * 8;
        float vj = silu(cji[r] + bjv);
        xji_out[(e0 + row) * 128 + col] = vj;
        float vk = silu(ckj[r] + bkv) * rbf_s[row * 132 + col];
        Abuf2[row * 136 + col] = (__bf16)vk;
    }
    __syncthreads();

    if (wave < 4) {  // down projection: [16,128] @ [128,64]
        floatx8 c = gemm_tile<128>(Abuf2, Wt_down + n0 * 128, lane);
#pragma unroll
        for (int r = 0; r < 8; ++r) {
            int row = r + half * 8;
            xkjd_out[(e0 + row) * 64 + col] = silu(c[r]);
        }
    }
}

// ---------------- triplet stage 1 (WMMA): hidden[T,8] = basis[T,K] @ W1[K,8] ----------
// K = real depth, KP = K padded to mult of 32, KPS = LDS row stride (bf16 elems).
// Block = 64 threads (2 waves), 32 triplets per block. N padded 8->16 (zero cols).
template<int K, int KP, int KPS>
__global__ void __launch_bounds__(64) k_hidden_wmma(const float* __restrict__ src,
                                                    const __bf16* __restrict__ W1p,
                                                    float* __restrict__ dst)
{
    __shared__ __bf16 rows[32 * KPS];
    int tid = threadIdx.x;
    int t0  = blockIdx.x * 32;

    // stage: 32*K contiguous floats -> bf16 LDS (row stride KPS), coalesced
    const float* base = src + (size_t)t0 * K;
    for (int i = tid; i < 32 * K; i += 64) {
        int row = i / K, k = i - row * K;
        rows[row * KPS + k] = (__bf16)base[i];
    }
    // zero the K..KP pad region
    for (int i = tid; i < 32 * (KP - K); i += 64) {
        int row = i / (KP - K), k = K + (i - row * (KP - K));
        rows[row * KPS + k] = (__bf16)0.f;
    }
    __syncthreads();

    int wave = tid >> 5, lane = tid & 31;
    int rc = lane & 15, half = lane >> 4;
    const __bf16* Atile = rows + wave * 16 * KPS;

    floatx8 c = {};
#pragma unroll
    for (int kc = 0; kc < KP; kc += 32) {
        bf16x16 a = load_frag(Atile, KPS, kc, lane);
        bf16x16 b = load_frag(W1p, KP, kc, lane);
        c = wmma_bf16(a, b, c);
    }
    if (rc < 8) {
#pragma unroll
        for (int r = 0; r < 8; ++r) {
            int row = r + half * 8;
            dst[(size_t)(t0 + wave * 16 + row) * 8 + rc] = c[r];
        }
    }
}

// ---------------- triplet stage 2: msg product + segment-sum ----------------
__global__ void __launch_bounds__(256) k_triplet(
    const float* __restrict__ Hs, const float* __restrict__ Ht,
    const float* __restrict__ W2t_s, const float* __restrict__ W2t_t,  // [64][8] c-major
    const float* __restrict__ xkjd,
    const int* __restrict__ idx_kj, const int* __restrict__ idx_ji,
    float* __restrict__ agg)
{
    int g = blockIdx.x * 256 + threadIdx.x;
    int t = g >> 6, c = g & 63;
    int kj = idx_kj[t], ji = idx_ji[t];

    const float4* hs = (const float4*)(Hs + t * 8);
    const float4* ht = (const float4*)(Ht + t * 8);
    const float4* ww = (const float4*)(W2t_s + c * 8);
    const float4* wv = (const float4*)(W2t_t + c * 8);
    float4 h0 = hs[0], h1 = hs[1], w0 = ww[0], w1 = ww[1];
    float s = h0.x * w0.x + h0.y * w0.y + h0.z * w0.z + h0.w * w0.w
            + h1.x * w1.x + h1.y * w1.y + h1.z * w1.z + h1.w * w1.w;
    float4 g0 = ht[0], g1 = ht[1], v0 = wv[0], v1 = wv[1];
    float p = g0.x * v0.x + g0.y * v0.y + g0.z * v0.z + g0.w * v0.w
            + g1.x * v1.x + g1.y * v1.y + g1.z * v1.z + g1.w * v1.w;

    float m = xkjd[kj * 64 + c] * s * p;
    atomicAdd(agg + ji * 64 + c, m);   // global_atomic_add_f32 (no return)
}

// ---------------- edge back: up + residual stack + lin + e2 ----------------
__global__ void __launch_bounds__(256) k_edge_back(
    const float* __restrict__ agg, const float* __restrict__ xji,
    const float* __restrict__ x1, const float* __restrict__ rbf0, const float* __restrict__ W_rbf,
    const __bf16* __restrict__ Wt_up,
    const __bf16* __restrict__ Wt_b1, const float* __restrict__ bb1,
    const __bf16* __restrict__ Wt_b2, const float* __restrict__ bb2,
    const __bf16* __restrict__ Wt_lin, const float* __restrict__ blin,
    const __bf16* __restrict__ Wt_a1, const float* __restrict__ ba1,
    const __bf16* __restrict__ Wt_a2, const float* __restrict__ ba2,
    float* __restrict__ out1, float* __restrict__ out2)
{
    __shared__ __bf16 Ab[16 * 136];
    __shared__ float  r0s[96];

    int tid = threadIdx.x;
    int e0  = blockIdx.x * 16;
    int wave = tid >> 5, lane = tid & 31;
    int n0 = wave * 16;
    int rc = lane & 15, half = lane >> 4;
    int col = n0 + rc;

    if (tid < 96) r0s[tid] = rbf0[e0 * 6 + tid];
    for (int i = tid; i < 16 * 64; i += 256) {
        int row = i >> 6, k = i & 63;
        Ab[row * 136 + k] = (__bf16)agg[(e0 + row) * 64 + k];
    }
    __syncthreads();

    float e1v[8];   // running e1 cells owned by this lane (fixed across stages)
    floatx8 c;

    // S1: e1 = x_ji + silu(agg @ W_up)          [K=64]
    c = gemm_tile<64>(Ab, Wt_up + n0 * 64, lane);
    __syncthreads();
#pragma unroll
    for (int r = 0; r < 8; ++r) {
        int row = r + half * 8;
        e1v[r] = xji[(e0 + row) * 128 + col] + silu(c[r]);
        Ab[row * 136 + col] = (__bf16)e1v[r];
    }
    __syncthreads();

    // S2/S3: res_before:  e1 += silu(silu(e1@W1+b1)@W2+b2)
    c = gemm_tile<128>(Ab, Wt_b1 + n0 * 128, lane);
    __syncthreads();
    {
        float bv = bb1[col];
#pragma unroll
        for (int r = 0; r < 8; ++r) {
            int row = r + half * 8;
            Ab[row * 136 + col] = (__bf16)silu(c[r] + bv);
        }
    }
    __syncthreads();
    c = gemm_tile<128>(Ab, Wt_b2 + n0 * 128, lane);
    __syncthreads();
    {
        float bv = bb2[col];
#pragma unroll
        for (int r = 0; r < 8; ++r) {
            int row = r + half * 8;
            e1v[r] += silu(c[r] + bv);
            Ab[row * 136 + col] = (__bf16)e1v[r];
        }
    }
    __syncthreads();

    // S4: e1 = silu(e1@W_lin + b_lin) + x1
    c = gemm_tile<128>(Ab, Wt_lin + n0 * 128, lane);
    __syncthreads();
    {
        float bv = blin[col];
#pragma unroll
        for (int r = 0; r < 8; ++r) {
            int row = r + half * 8;
            e1v[r] = silu(c[r] + bv) + x1[(e0 + row) * 128 + col];
            Ab[row * 136 + col] = (__bf16)e1v[r];
        }
    }
    __syncthreads();

    // S5..S8: two res_after blocks
#pragma unroll
    for (int k = 0; k < 2; ++k) {
        c = gemm_tile<128>(Ab, Wt_a1 + k * (128 * 128) + n0 * 128, lane);
        __syncthreads();
        {
            float bv = ba1[k * 128 + col];
#pragma unroll
            for (int r = 0; r < 8; ++r) {
                int row = r + half * 8;
                Ab[row * 136 + col] = (__bf16)silu(c[r] + bv);
            }
        }
        __syncthreads();
        c = gemm_tile<128>(Ab, Wt_a2 + k * (128 * 128) + n0 * 128, lane);
        __syncthreads();
        {
            float bv = ba2[k * 128 + col];
#pragma unroll
            for (int r = 0; r < 8; ++r) {
                int row = r + half * 8;
                e1v[r] += silu(c[r] + bv);
                Ab[row * 136 + col] = (__bf16)e1v[r];
            }
        }
        __syncthreads();
    }

    // epilogue: e1 out, e2 = (rbf0 @ W_rbf) * e1
#pragma unroll
    for (int r = 0; r < 8; ++r) {
        int row = r + half * 8;
        out1[(e0 + row) * 128 + col] = e1v[r];
        float rb = 0.f;
#pragma unroll
        for (int j = 0; j < 6; ++j) rb += r0s[row * 6 + j] * W_rbf[j * 128 + col];
        out2[(e0 + row) * 128 + col] = rb * e1v[r];
    }
}

// ---------------- host launch ----------------
extern "C" void kernel_launch(void* const* d_in, const int* in_sizes, int n_in,
                              void* d_out, int out_size, void* d_ws, size_t ws_size,
                              hipStream_t stream) {
    (void)in_sizes; (void)n_in; (void)out_size; (void)ws_size;

    const float* x1      = (const float*)d_in[0];
    const float* rbf0    = (const float*)d_in[1];
    const float* sbf     = (const float*)d_in[2];
    const float* t_basis = (const float*)d_in[3];
    const int*   idx_kj  = (const int*)d_in[4];
    const int*   idx_ji  = (const int*)d_in[5];
    const float* W_rbf1  = (const float*)d_in[6];
    const float* W_rbf2  = (const float*)d_in[7];
    const float* W_sbf1  = (const float*)d_in[8];
    const float* W_sbf2  = (const float*)d_in[9];
    const float* W_t1    = (const float*)d_in[10];
    const float* W_t2    = (const float*)d_in[11];
    const float* W_rbf   = (const float*)d_in[12];
    const float* W_kj    = (const float*)d_in[13];
    const float* b_kj    = (const float*)d_in[14];
    const float* W_ji    = (const float*)d_in[15];
    const float* b_ji    = (const float*)d_in[16];
    const float* W_down  = (const float*)d_in[17];
    const float* W_up    = (const float*)d_in[18];
    const float* rbW1    = (const float*)d_in[19];
    const float* rbb1    = (const float*)d_in[20];
    const float* rbW2    = (const float*)d_in[21];
    const float* rbb2    = (const float*)d_in[22];
    const float* W_lin   = (const float*)d_in[23];
    const float* b_lin   = (const float*)d_in[24];
    const float* raW1    = (const float*)d_in[25];
    const float* rab1    = (const float*)d_in[26];
    const float* raW2    = (const float*)d_in[27];
    const float* rab2    = (const float*)d_in[28];

    uint8_t* ws = (uint8_t*)d_ws;
    // bf16 weights (N-major) + fused rbf matrix + padded/transposed triplet weights
    __bf16* Wt_ji   = (__bf16*)(ws + 0);
    __bf16* Wt_kj   = (__bf16*)(ws + 32768);
    __bf16* Wt_b1   = (__bf16*)(ws + 65536);
    __bf16* Wt_b2   = (__bf16*)(ws + 98304);
    __bf16* Wt_lin  = (__bf16*)(ws + 131072);
    __bf16* Wt_a1   = (__bf16*)(ws + 163840);   // 2 slices
    __bf16* Wt_a2   = (__bf16*)(ws + 229376);   // 2 slices
    __bf16* Wt_down = (__bf16*)(ws + 294912);
    __bf16* Wt_up   = (__bf16*)(ws + 311296);
    float*  Mrbf    = (float*)(ws + 327680);    // 3072 B
    __bf16* W1p_s   = (__bf16*)(ws + 330752);   // [16 x 64]  bf16
    __bf16* W1p_t   = (__bf16*)(ws + 332800);   // [16 x 320] bf16
    float*  W2t_s   = (float*)(ws + 343040);    // [64 x 8] f32
    float*  W2t_t   = (float*)(ws + 345088);    // [64 x 8] f32
    float*  xji     = (float*)(ws + 347136);           // E*128
    float*  xkjd    = (float*)(ws + 51547136ull);      // E*64
    float*  Hs      = (float*)(ws + 77147136ull);      // T*8
    float*  Ht      = (float*)(ws + 93147136ull);      // T*8
    float*  agg     = (float*)(ws + 109147136ull);     // E*64

    float* out1 = (float*)d_out;
    float* out2 = out1 + (size_t)EN * HD;

    // weight prep
    k_transpose_bf16<<<64, 256, 0, stream>>>(W_ji,  Wt_ji, 128, 128);
    k_transpose_bf16<<<64, 256, 0, stream>>>(W_kj,  Wt_kj, 128, 128);
    k_transpose_bf16<<<64, 256, 0, stream>>>(rbW1,  Wt_b1, 128, 128);
    k_transpose_bf16<<<64, 256, 0, stream>>>(rbW2,  Wt_b2, 128, 128);
    k_transpose_bf16<<<64, 256, 0, stream>>>(W_lin, Wt_lin, 128, 128);
    k_transpose_bf16<<<64, 256, 0, stream>>>(raW1,          Wt_a1,         128, 128);
    k_transpose_bf16<<<64, 256, 0, stream>>>(raW1 + 16384,  Wt_a1 + 16384, 128, 128);
    k_transpose_bf16<<<64, 256, 0, stream>>>(raW2,          Wt_a2,         128, 128);
    k_transpose_bf16<<<64, 256, 0, stream>>>(raW2 + 16384,  Wt_a2 + 16384, 128, 128);
    k_transpose_bf16<<<32, 256, 0, stream>>>(W_down, Wt_down, 128, 64);
    k_transpose_bf16<<<32, 256, 0, stream>>>(W_up,   Wt_up,   64, 128);
    k_mrbf<<<1, 256, 0, stream>>>(W_rbf1, W_rbf2, Mrbf);
    k_pad_w1<<<4,  256, 0, stream>>>(W_sbf1, W1p_s, SBF_K, 64);
    k_pad_w1<<<20, 256, 0, stream>>>(W_t1,   W1p_t, TB_K, 320);
    k_transpose_f32<<<2, 256, 0, stream>>>(W_sbf2, W2t_s);
    k_transpose_f32<<<2, 256, 0, stream>>>(W_t2,   W2t_t);

    // edge front: x_ji, x_kj_down
    k_edge_front<<<EN / 16, 256, 0, stream>>>(x1, rbf0, Mrbf, Wt_ji, b_ji, Wt_kj, b_kj,
                                              Wt_down, xji, xkjd);

    // triplet pipeline (WMMA basis projections, 32 triplets / 2-wave block)
    k_hidden_wmma<SBF_K, 64, 72><<<TN / 32, 64, 0, stream>>>(sbf, W1p_s, Hs);
    k_hidden_wmma<TB_K, 320, 328><<<TN / 32, 64, 0, stream>>>(t_basis, W1p_t, Ht);
    k_zero<<<(EN * ID) / 256, 256, 0, stream>>>(agg, EN * ID);
    k_triplet<<<(TN * ID) / 256, 256, 0, stream>>>(Hs, Ht, W2t_s, W2t_t, xkjd,
                                                   idx_kj, idx_ji, agg);

    // edge back: up + residual stack + e2
    k_edge_back<<<EN / 16, 256, 0, stream>>>(agg, xji, x1, rbf0, W_rbf, Wt_up,
                                             Wt_b1, rbb1, Wt_b2, rbb2, Wt_lin, b_lin,
                                             Wt_a1, rab1, Wt_a2, rab2, out1, out2);
}
